// MICF_Net_40510131536241
// MI455X (gfx1250) — matrix operand order repaired and checked by reading, not compiled, not run on hardware
//
#include <hip/hip_runtime.h>
#include <hip/hip_bf16.h>

typedef __attribute__((ext_vector_type(16))) _Float16 v16h;
typedef __attribute__((ext_vector_type(8)))  _Float16 v8h;
typedef __attribute__((ext_vector_type(8)))  float    v8f;
typedef _Float16 half_t;

#define BATCH 128
#define NTOK  122
#define TOKENS (BATCH*NTOK)   // 15616
#define SP    121
#define MSP   (BATCH*SP)      // 15488
#define DIMC  64
#define HC    8
#define HD    8
#define NCLS  15
#define C2    1120
#define PERB  (HC*NTOK*NTOK)  // 119072
#define SCALE_ATT 0.35355339059327373f
#define BN_RS 0.9999950000374996f   // rsqrt(1+1e-5)

// ---------------- generic WMMA GEMM, 16M x 64N per wave, double-buffered K pipeline ------------
// C[M,Nreal] = res + act(A[M,Kpad] * W[Npad,Kpad]^T + bias);  act: 0=none, 1=bn+leaky, 2=gelu
// K-step k+1 loads are issued before the 4 WMMAs of step k consume their buffers, so the
// matrix ops overlap the 10 in-flight global loads (loads return in order -> partial waits).
__global__ void gemm_wmma(const half_t* __restrict__ A, const half_t* __restrict__ W,
                          const float* __restrict__ bias, const float* __restrict__ res,
                          float* __restrict__ C, int M, int Nreal, int Npad, int Kpad,
                          int lda, int ldc, int act,
                          const float* __restrict__ g, const float* __restrict__ beta)
{
    const int lane = threadIdx.x;
    const int mt = blockIdx.x * 16, nt0 = blockIdx.y * 64;
    const int ml = lane & 15, hiH = lane >> 4;
    const half_t* Arow = A + (size_t)(mt + ml) * lda + (hiH ? 8 : 0);
    const half_t* Wrow[4];
#pragma unroll
    for (int nb = 0; nb < 4; ++nb) {
        int wr = nt0 + nb * 16 + ml;
        if (wr >= Npad) wr = Npad - 1;          // clamp: garbage accumulated, never stored
        Wrow[nb] = W + (size_t)wr * Kpad + (hiH ? 8 : 0);
    }
    v8f acc[4] = {};
    // ---- prologue: load K-step 0 into buffer 0 ----
    v8h alo0 = *(const v8h*)(Arow);
    v8h ahi0 = *(const v8h*)(Arow + 16);
    v8h blo0[4], bhi0[4];
#pragma unroll
    for (int nb = 0; nb < 4; ++nb) {
        blo0[nb] = *(const v8h*)(Wrow[nb]);
        bhi0[nb] = *(const v8h*)(Wrow[nb] + 16);
    }
    for (int kt = 0; kt < Kpad; kt += 32) {
        // ---- prefetch next K-step into buffer 1 (clamped on last iteration) ----
        const int kn = (kt + 32 < Kpad) ? kt + 32 : kt;
        v8h alo1 = *(const v8h*)(Arow + kn);
        v8h ahi1 = *(const v8h*)(Arow + kn + 16);
        v8h blo1[4], bhi1[4];
#pragma unroll
        for (int nb = 0; nb < 4; ++nb) {
            blo1[nb] = *(const v8h*)(Wrow[nb] + kn);
            bhi1[nb] = *(const v8h*)(Wrow[nb] + kn + 16);
        }
        // ---- compute on buffer 0: 4 independent WMMA chains ----
        v16h a;
#pragma unroll
        for (int i = 0; i < 8; ++i) { a[i] = alo0[i]; a[8+i] = ahi0[i]; }
#pragma unroll
        for (int nb = 0; nb < 4; ++nb) {
            v16h b;
#pragma unroll
            for (int i = 0; i < 8; ++i) { b[i] = blo0[nb][i]; b[8+i] = bhi0[nb][i]; }
            acc[nb] = __builtin_amdgcn_wmma_f32_16x16x32_f16(false, a, false, b, (short)0, acc[nb], false, false);
        }
        // ---- rotate buffers ----
        alo0 = alo1; ahi0 = ahi1;
#pragma unroll
        for (int nb = 0; nb < 4; ++nb) { blo0[nb] = blo1[nb]; bhi0[nb] = bhi1[nb]; }
    }
    const int mb = mt + (hiH ? 8 : 0);
#pragma unroll
    for (int nb = 0; nb < 4; ++nb) {
        const int n = nt0 + nb * 16 + ml;
        if (n >= Nreal) continue;
#pragma unroll
        for (int r = 0; r < 8; ++r) {
            int m = mb + r;
            if (m < M) {
                float v = acc[nb][r] + bias[n];
                if (act == 1)      { v = v * (g[n] * BN_RS) + beta[n]; v = (v >= 0.f) ? v : 0.01f * v; }
                else if (act == 2) { v = 0.5f * v * (1.f + erff(v * 0.70710678118654752f)); }
                if (res) v += res[(size_t)m * ldc + n];
                C[(size_t)m * ldc + n] = v;
            }
        }
    }
}

// ---------------- attention scores: dots[bh,i,j] = SCALE * q_i . k_j (K=8 zero padded) ----------
__global__ void attn_scores(const float* __restrict__ qkv, float* __restrict__ dots)
{
    const int lane = threadIdx.x;
    const int it = blockIdx.x, jt = blockIdx.y, bh = blockIdx.z;
    const int b = bh >> 3, h = bh & 7;
    const int ml = lane & 15, hiH = lane >> 4;
    v16h a = {}, bb = {};
    if (!hiH) {
        int i = it * 16 + ml; if (i > NTOK - 1) i = NTOK - 1;
        const float* qp = qkv + ((size_t)(b * NTOK + i)) * 192 + h * 8;
        int j = jt * 16 + ml; if (j > NTOK - 1) j = NTOK - 1;
        const float* kp = qkv + ((size_t)(b * NTOK + j)) * 192 + 64 + h * 8;
#pragma unroll
        for (int d = 0; d < 8; ++d) { a[d] = (half_t)qp[d]; bb[d] = (half_t)kp[d]; }
    }
    v8f acc = {};
    acc = __builtin_amdgcn_wmma_f32_16x16x32_f16(false, a, false, bb, (short)0, acc, false, false);
    const int j = jt * 16 + ml;
    const int ib = it * 16 + (hiH ? 8 : 0);
    float* dbase = dots + (size_t)bh * NTOK * NTOK;
#pragma unroll
    for (int r = 0; r < 8; ++r) {
        int i = ib + r;
        if (i < NTOK && j < NTOK) dbase[(size_t)i * NTOK + j] = acc[r] * SCALE_ATT;
    }
}

// ---------------- P(16x128 f16) x V(128x8) WMMA, ctx[b,i,h*8+d] -------------------------------
__global__ void attn_pv(const half_t* __restrict__ P, const float* __restrict__ qkv,
                        float* __restrict__ ctx)
{
    const int lane = threadIdx.x;
    const int it = blockIdx.x, bh = blockIdx.y;
    const int b = bh >> 3, h = bh & 7;
    const int ml = lane & 15, hiH = lane >> 4;
    int i = it * 16 + ml; if (i > NTOK - 1) i = NTOK - 1;
    const half_t* Prow = P + ((size_t)bh * 128 + i) * 128 + (hiH ? 8 : 0);
    v8f acc = {};
#pragma unroll
    for (int kt = 0; kt < 128; kt += 32) {
        v8h alo = *(const v8h*)(Prow + kt);
        v8h ahi = *(const v8h*)(Prow + kt + 16);
        v16h a, bb = {};
#pragma unroll
        for (int t = 0; t < 8; ++t) { a[t] = alo[t]; a[8+t] = ahi[t]; }
        if (ml < 8) {
#pragma unroll
            for (int t = 0; t < 16; ++t) {
                int j = kt + (hiH ? 8 : 0) + (t < 8 ? t : t + 8); if (j > NTOK - 1) j = NTOK - 1;
                bb[t] = (half_t)qkv[((size_t)(b * NTOK + j)) * 192 + 128 + h * 8 + ml];
            }
        }
        acc = __builtin_amdgcn_wmma_f32_16x16x32_f16(false, a, false, bb, (short)0, acc, false, false);
    }
    const int ib = it * 16 + (hiH ? 8 : 0);
#pragma unroll
    for (int r = 0; r < 8; ++r) {
        int i2 = ib + r;
        if (i2 < NTOK && ml < 8) ctx[((size_t)(b * NTOK + i2)) * 64 + h * 8 + ml] = acc[r];
    }
}

// ---------------- pointwise / glue kernels ----------------------------------------------------
__global__ void conv3d_h2(const float* __restrict__ x, const float* __restrict__ w3d,
                          const float* __restrict__ b3d, const float* __restrict__ bn,
                          half_t* __restrict__ h2, int total)
{
    int idx = blockIdx.x * blockDim.x + threadIdx.x;
    if (idx >= total) return;
    int c = idx % C2, rp = idx / C2, p = rp % SP, b = rp / SP;
    int o = c / 140, dd = c % 140;
    const float* xp = x + ((size_t)b * 144 + dd) * SP + p;
    float acc = b3d[o];
#pragma unroll
    for (int k = 0; k < 5; ++k) acc += xp[(size_t)k * SP] * w3d[o * 5 + k];
    float v = acc * (bn[o] * BN_RS) + bn[8 + o];
    h2[idx] = (half_t)((v >= 0.f) ? v : 0.01f * v);
}

__global__ void lidar_pack(const float* __restrict__ lidar, half_t* __restrict__ A, int total)
{
    int idx = blockIdx.x * blockDim.x + threadIdx.x;
    if (idx >= total) return;
    int c = idx & 31, rp = idx >> 5, p = rp % SP, b = rp / SP;
    A[idx] = (c < 4) ? (half_t)lidar[((size_t)b * 4 + c) * SP + p] : (half_t)0.f;
}

__global__ void pad2d_f16(const float* __restrict__ src, half_t* __restrict__ dst,
                          int R, int C, int Rout, int Cout)
{
    int idx = blockIdx.x * blockDim.x + threadIdx.x;
    if (idx >= Rout * Cout) return;
    int c = idx % Cout, r = idx / Cout;
    dst[idx] = (r < R && c < C) ? (half_t)src[(size_t)r * C + c] : (half_t)0.f;
}

__global__ void build_tokens(const float* __restrict__ feat, const float* __restrict__ cls,
                             const float* __restrict__ pos, float* __restrict__ tok, int total)
{
    int idx = blockIdx.x * blockDim.x + threadIdx.x;
    if (idx >= total) return;
    int d = idx & 63, rem = idx >> 6, n = rem % NTOK, b = rem / NTOK;
    float v = (n == 0) ? cls[d] : feat[((size_t)b * SP + (n - 1)) * 64 + d];
    tok[idx] = v + pos[n * 64 + d];
}

__global__ void layernorm_f16(const float* __restrict__ x, const float* __restrict__ g,
                              const float* __restrict__ b_, half_t* __restrict__ out)
{
    int tok = blockIdx.x, d = threadIdx.x;
    float v = x[(size_t)tok * 64 + d];
    __shared__ float red[64];
    red[d] = v; __syncthreads();
    for (int o = 32; o > 0; o >>= 1) { if (d < o) red[d] += red[d + o]; __syncthreads(); }
    float mu = red[0] * (1.f / 64.f); __syncthreads();
    float c = v - mu;
    red[d] = c * c; __syncthreads();
    for (int o = 32; o > 0; o >>= 1) { if (d < o) red[d] += red[d + o]; __syncthreads(); }
    float var = red[0] * (1.f / 64.f);
    out[(size_t)tok * 64 + d] = (half_t)(c * rsqrtf(var + 1e-5f) * g[d] + b_[d]);
}

__global__ void sup_mix(float* __restrict__ dx, float* __restrict__ dl,
                        const float* __restrict__ w1, const float* __restrict__ b1,
                        const float* __restrict__ bng, const float* __restrict__ bnb,
                        const float* __restrict__ w2, const float* __restrict__ b2, int total)
{
    int idx = blockIdx.x * blockDim.x + threadIdx.x;
    if (idx >= total) return;
    int h = (idx / (NTOK * NTOK)) & 7;
    float d0 = dx[idx], d1 = dl[idx];
    float sup = b2[h];
#pragma unroll
    for (int a = 0; a < 8; ++a) {
        float t = d0 * w1[(h * 8 + a) * 2] + d1 * w1[(h * 8 + a) * 2 + 1] + b1[h * 8 + a];
        t = t * (bng[h * 8 + a] * BN_RS) + bnb[h * 8 + a];
        t = (t >= 0.f) ? t : 0.01f * t;
        sup += t * w2[h * 8 + a];
    }
    dx[idx] = d0 + sup;
    dl[idx] = d1 + sup;
}

__global__ void kl_max(const float* __restrict__ dx, const float* __restrict__ dl,
                       float* __restrict__ mx, float* __restrict__ ml)
{
    int b = blockIdx.x, t = threadIdx.x;
    const float* px = dx + (size_t)b * PERB;
    const float* pl = dl + (size_t)b * PERB;
    float m0 = -3.4e38f, m1 = -3.4e38f;
    for (int i = t; i < PERB; i += 256) { m0 = fmaxf(m0, px[i]); m1 = fmaxf(m1, pl[i]); }
    __shared__ float s0[256], s1[256];
    s0[t] = m0; s1[t] = m1; __syncthreads();
    for (int o = 128; o > 0; o >>= 1) { if (t < o) { s0[t] = fmaxf(s0[t], s0[t+o]); s1[t] = fmaxf(s1[t], s1[t+o]); } __syncthreads(); }
    if (!t) { mx[b] = s0[0]; ml[b] = s1[0]; }
}

__global__ void kl_sumexp(const float* __restrict__ dx, const float* __restrict__ dl,
                          const float* __restrict__ mx, const float* __restrict__ ml,
                          float* __restrict__ sx, float* __restrict__ sl)
{
    int b = blockIdx.x, t = threadIdx.x;
    const float* px = dx + (size_t)b * PERB;
    const float* pl = dl + (size_t)b * PERB;
    float m0 = mx[b], m1 = ml[b], a0 = 0.f, a1 = 0.f;
    for (int i = t; i < PERB; i += 256) { a0 += __expf(px[i] - m0); a1 += __expf(pl[i] - m1); }
    __shared__ float s0[256], s1[256];
    s0[t] = a0; s1[t] = a1; __syncthreads();
    for (int o = 128; o > 0; o >>= 1) { if (t < o) { s0[t] += s0[t+o]; s1[t] += s1[t+o]; } __syncthreads(); }
    if (!t) { sx[b] = s0[0]; sl[b] = s1[0]; }
}

__global__ void kl_accum(const float* __restrict__ dx, const float* __restrict__ dl,
                         const float* __restrict__ mx, const float* __restrict__ ml,
                         const float* __restrict__ sx, const float* __restrict__ sl,
                         float* __restrict__ scal)
{
    int b = blockIdx.x, t = threadIdx.x;
    const float* px = dx + (size_t)b * PERB;
    const float* pl = dl + (size_t)b * PERB;
    float offp = mx[b] + logf(sx[b]);
    float offq = ml[b] + logf(sl[b]);
    float acc = 0.f;
    for (int i = t; i < PERB; i += 256) {
        float lq = pl[i] - offq, lp = px[i] - offp;
        acc += __expf(lq) * (lq - lp);
    }
    __shared__ float s0[256];
    s0[t] = acc; __syncthreads();
    for (int o = 128; o > 0; o >>= 1) { if (t < o) s0[t] += s0[t+o]; __syncthreads(); }
    if (!t) atomicAdd(&scal[2], s0[0] * (1.f / BATCH));
}

__global__ void softmax_row(const float* __restrict__ dots, half_t* __restrict__ P)
{
    int row = blockIdx.x;                  // bh*122 + i
    int bh = row / NTOK, i = row % NTOK;
    int lane = threadIdx.x;
    const float* d = dots + (size_t)row * NTOK;
    float mx = -3.4e38f;
    for (int j = lane; j < NTOK; j += 32) mx = fmaxf(mx, d[j]);
    for (int o = 16; o > 0; o >>= 1) mx = fmaxf(mx, __shfl_xor(mx, o, 32));
    float s = 0.f;
    for (int j = lane; j < NTOK; j += 32) s += __expf(d[j] - mx);
    for (int o = 16; o > 0; o >>= 1) s += __shfl_xor(s, o, 32);
    float inv = 1.f / s;
    half_t* pr = P + ((size_t)bh * 128 + i) * 128;
    for (int j = lane; j < 128; j += 32)
        pr[j] = (j < NTOK) ? (half_t)(__expf(d[j] - mx) * inv) : (half_t)0.f;
}

__global__ void proto_scores(const float* __restrict__ xtok, const float* __restrict__ ltok,
                             const float* __restrict__ xp, const float* __restrict__ lp,
                             float* __restrict__ s1, float* __restrict__ s2,
                             half_t* __restrict__ s1p, half_t* __restrict__ s2p)
{
    int b = blockIdx.x, c = threadIdx.x;
    __shared__ float v1[16], v2[16];
    if (c < NCLS) {
        float d2a = 0.f, d2b = 0.f;
        for (int d = 0; d < 64; ++d) {
            float ta = xtok[((size_t)b * NTOK) * 64 + d] - xp[c * 64 + d];
            float tb = ltok[((size_t)b * NTOK) * 64 + d] - lp[c * 64 + d];
            d2a += ta * ta; d2b += tb * tb;
        }
        v1[c] = -sqrtf(d2a + 1e-12f);
        v2[c] = -sqrtf(d2b + 1e-12f);
    }
    __syncthreads();
    if (c < NCLS) {
        float m1 = -3.4e38f, m2 = -3.4e38f;
        for (int k = 0; k < NCLS; ++k) { m1 = fmaxf(m1, v1[k]); m2 = fmaxf(m2, v2[k]); }
        float z1 = 0.f, z2 = 0.f;
        for (int k = 0; k < NCLS; ++k) { z1 += __expf(v1[k] - m1); z2 += __expf(v2[k] - m2); }
        float r1 = __expf(v1[c] - m1) / z1, r2 = __expf(v2[c] - m2) / z2;
        s1[b * NCLS + c] = r1; s2[b * NCLS + c] = r2;
        s1p[b * 32 + c] = (half_t)r1; s2p[b * 32 + c] = (half_t)r2;
    } else {
        s1p[b * 32 + c] = (half_t)0.f; s2p[b * 32 + c] = (half_t)0.f;
    }
}

__global__ void pro_kernel(const float* __restrict__ s1, const float* __restrict__ s2,
                           const int* __restrict__ label, float* __restrict__ scal)
{
    int b = threadIdx.x;
    __shared__ float a1[128], a2[128];
    float m1 = -3.4e38f, m2 = -3.4e38f;
    for (int c = 0; c < NCLS; ++c) { m1 = fmaxf(m1, s1[b*NCLS+c]); m2 = fmaxf(m2, s2[b*NCLS+c]); }
    float z1 = 0.f, z2 = 0.f;
    for (int c = 0; c < NCLS; ++c) { z1 += __expf(s1[b*NCLS+c]-m1); z2 += __expf(s2[b*NCLS+c]-m2); }
    int lab = label[b];
    a1[b] = __expf(s1[b*NCLS+lab]-m1) / z1;
    a2[b] = __expf(s2[b*NCLS+lab]-m2) / z2;
    __syncthreads();
    for (int o = 64; o > 0; o >>= 1) { if (b < o) { a1[b] += a1[b+o]; a2[b] += a2[b+o]; } __syncthreads(); }
    if (!b) {
        float r1 = a1[0] / a2[0];
        scal[0] = (r1 > 1.f) ? 1.f - tanhf(0.1f * r1) : 1.f;
        scal[1] = (r1 > 1.f) ? 1.f : 1.f - tanhf(0.1f / r1);
    }
}

__global__ void mask_mul(const float* __restrict__ t, const float* __restrict__ rnd,
                         const float* __restrict__ scal, int which,
                         float* __restrict__ out, half_t* __restrict__ out16, int total)
{
    int idx = blockIdx.x * blockDim.x + threadIdx.x;
    if (idx >= total) return;
    int d = idx & 63;
    float p = scal[which];
    float f = (rnd[d] <= p) ? 1.f : p;
    float v = t[idx] * f;
    out[idx] = v; out16[idx] = (half_t)v;
}

__global__ void fuse_attn(const float* __restrict__ q, const float* __restrict__ k,
                          const float* __restrict__ v, float* __restrict__ out)
{
    int bh = blockIdx.x, b = bh >> 3, h = bh & 7;
    int t = threadIdx.x;
    __shared__ float s[128], red[128];
    float qv[8];
#pragma unroll
    for (int d = 0; d < 8; ++d) qv[d] = q[b * 64 + h * 8 + d];
    float val = -3.4e38f;
    if (t < NTOK) {
        float dot = 0.f;
        const float* kp = k + ((size_t)(b * NTOK + t)) * 64 + h * 8;
#pragma unroll
        for (int d = 0; d < 8; ++d) dot += qv[d] * kp[d];
        val = dot * SCALE_ATT;
    }
    s[t] = val; red[t] = val; __syncthreads();
    for (int o = 64; o > 0; o >>= 1) { if (t < o) red[t] = fmaxf(red[t], red[t+o]); __syncthreads(); }
    float mx = red[0]; __syncthreads();
    float e = (t < NTOK) ? __expf(s[t] - mx) : 0.f;
    s[t] = e; red[t] = e; __syncthreads();
    for (int o = 64; o > 0; o >>= 1) { if (t < o) red[t] += red[t+o]; __syncthreads(); }
    float inv = 1.f / red[0];
    if (t < 8) {
        float acc = 0.f;
        for (int j = 0; j < NTOK; ++j) acc += s[j] * v[((size_t)(b * NTOK + j)) * 64 + h * 8 + t];
        out[b * 64 + h * 8 + t] = acc * inv;
    }
}

__global__ void build_fused(const float* __restrict__ f1, const float* __restrict__ f2,
                            const float* __restrict__ xm, const float* __restrict__ lm,
                            float* __restrict__ fused, half_t* __restrict__ fused16)
{
    int idx = blockIdx.x * blockDim.x + threadIdx.x;
    if (idx >= BATCH * 128) return;
    int d = idx & 127, b = idx >> 7;
    float v;
    if (d < 64) v = f1[b * 64 + d] + xm[((size_t)b * NTOK) * 64 + d];
    else { int dd = d - 64; v = f2[b * 64 + dd] + lm[((size_t)b * NTOK) * 64 + dd]; }
    fused[idx] = v; fused16[idx] = (half_t)v;
}

__global__ void copy_out(const float* __restrict__ scal, const float* __restrict__ xtok,
                         const float* __restrict__ ltok, float* __restrict__ out)
{
    int idx = blockIdx.x * blockDim.x + threadIdx.x;
    if (idx == 0) out[1920] = scal[2];
    if (idx >= 1 && idx <= 8192) {
        int i = idx - 1; int b = i >> 6, d = i & 63;
        out[1921 + i] = xtok[((size_t)b * NTOK) * 64 + d];
    } else if (idx > 8192 && idx <= 16384) {
        int i = idx - 8193; int b = i >> 6, d = i & 63;
        out[1921 + 8192 + i] = ltok[((size_t)b * NTOK) * 64 + d];
    }
}

__global__ void init_scal(float* scal) { if (threadIdx.x < 16) scal[threadIdx.x] = 0.f; }

// ======================================================================================
extern "C" void kernel_launch(void* const* d_in, const int* in_sizes, int n_in,
                              void* d_out, int out_size, void* d_ws, size_t ws_size,
                              hipStream_t stream)
{
    (void)in_sizes; (void)n_in; (void)out_size; (void)ws_size;
    const float* x      = (const float*)d_in[0];
    const float* lidar  = (const float*)d_in[1];
    const float* x_proto= (const float*)d_in[2];
    const float* l_proto= (const float*)d_in[3];
    const float* rand1  = (const float*)d_in[4];
    const float* rand2  = (const float*)d_in[5];
    const float* w3d    = (const float*)d_in[6];
    const float* b3d    = (const float*)d_in[7];
    const float* bn3d   = (const float*)d_in[8];
    const float* convx_w= (const float*)d_in[9];
    const float* convx_b= (const float*)d_in[10];
    const float* bnx    = (const float*)d_in[11];
    const float* lidw1  = (const float*)d_in[12];
    const float* lidb1  = (const float*)d_in[13];
    const float* bnl1   = (const float*)d_in[14];
    const float* lidw2  = (const float*)d_in[15];
    const float* lidb2  = (const float*)d_in[16];
    const float* bnl2   = (const float*)d_in[17];
    const float* cls_tok= (const float*)d_in[18];
    const float* pos1   = (const float*)d_in[19];
    const float* pos2   = (const float*)d_in[20];
    const float* ln_att = (const float*)d_in[21];
    const float* qkv_w  = (const float*)d_in[22];
    const float* qkv_b  = (const float*)d_in[23];
    const float* aw1    = (const float*)d_in[24];
    const float* ab1    = (const float*)d_in[25];
    const float* abn    = (const float*)d_in[26];
    const float* aw2    = (const float*)d_in[27];
    const float* ab2    = (const float*)d_in[28];
    const float* enc_mw = (const float*)d_in[29];
    const float* enc_mb = (const float*)d_in[30];
    const float* ln_ff  = (const float*)d_in[31];
    const float* ff_w1  = (const float*)d_in[32];
    const float* ff_b1  = (const float*)d_in[33];
    const float* ff_w2  = (const float*)d_in[34];
    const float* ff_b2  = (const float*)d_in[35];
    const float* fq_w   = (const float*)d_in[36];
    const float* fq_b   = (const float*)d_in[37];
    const float* fkv_w  = (const float*)d_in[38];
    const float* fkv_b  = (const float*)d_in[39];
    const float* head_w = (const float*)d_in[40];
    const float* head_b = (const float*)d_in[41];
    const int*   label  = (const int*)d_in[42];

    char* base = (char*)d_ws;
    size_t off = 0;
    auto alloc = [&](size_t bytes) -> char* {
        char* p = base + off;
        off += (bytes + 255) & ~(size_t)255;
        return p;
    };
    half_t* h2f16   = (half_t*)alloc((size_t)MSP * C2 * 2);
    half_t* lidA16  = (half_t*)alloc((size_t)MSP * 32 * 2);
    float*  lf1     = (float*) alloc((size_t)MSP * 32 * 4);
    half_t* lf1_16  = (half_t*)alloc((size_t)MSP * 32 * 2);
    float*  xfeat   = (float*) alloc((size_t)MSP * 64 * 4);
    float*  lfeat   = (float*) alloc((size_t)MSP * 64 * 4);
    float*  xtok    = (float*) alloc((size_t)TOKENS * 64 * 4);
    float*  ltok    = (float*) alloc((size_t)TOKENS * 64 * 4);
    half_t* ln16    = (half_t*)alloc((size_t)TOKENS * 64 * 2);
    float*  qkvx    = (float*) alloc((size_t)TOKENS * 192 * 4);
    float*  qkvl    = (float*) alloc((size_t)TOKENS * 192 * 4);
    float*  dots_x  = (float*) alloc((size_t)BATCH * PERB * 4);
    float*  dots_l  = (float*) alloc((size_t)BATCH * PERB * 4);
    half_t* P16     = (half_t*)alloc((size_t)BATCH * HC * 128 * 128 * 2);
    float*  ctx     = (float*) alloc((size_t)TOKENS * 64 * 4);
    half_t* ctx16   = (half_t*)alloc((size_t)TOKENS * 64 * 2);
    float*  hidden  = (float*) alloc((size_t)TOKENS * 8 * 4);
    half_t* hid16   = (half_t*)alloc((size_t)TOKENS * 32 * 2);
    float*  xm      = (float*) alloc((size_t)TOKENS * 64 * 4);
    float*  lm      = (float*) alloc((size_t)TOKENS * 64 * 4);
    half_t* xm16    = (half_t*)alloc((size_t)TOKENS * 64 * 2);
    half_t* lm16    = (half_t*)alloc((size_t)TOKENS * 64 * 2);
    float*  kbuf    = (float*) alloc((size_t)TOKENS * 64 * 4);
    float*  vbuf    = (float*) alloc((size_t)TOKENS * 64 * 4);
    float*  q1      = (float*) alloc(BATCH * 64 * 4);
    float*  q2      = (float*) alloc(BATCH * 64 * 4);
    float*  f1      = (float*) alloc(BATCH * 64 * 4);
    float*  f2      = (float*) alloc(BATCH * 64 * 4);
    float*  score1  = (float*) alloc(BATCH * NCLS * 4);
    float*  score2  = (float*) alloc(BATCH * NCLS * 4);
    half_t* s1p     = (half_t*)alloc(BATCH * 32 * 2);
    half_t* s2p     = (half_t*)alloc(BATCH * 32 * 2);
    float*  fused   = (float*) alloc(BATCH * 128 * 4);
    half_t* fused16 = (half_t*)alloc(BATCH * 128 * 2);
    half_t* convxw16= (half_t*)alloc((size_t)64 * C2 * 2);
    half_t* lidw1_16= (half_t*)alloc(32 * 32 * 2);
    half_t* lidw2_16= (half_t*)alloc(64 * 32 * 2);
    half_t* qkvw16  = (half_t*)alloc(192 * 64 * 2);
    half_t* encw16  = (half_t*)alloc(64 * 64 * 2);
    half_t* ffw1_16 = (half_t*)alloc(16 * 64 * 2);
    half_t* ffw2_16 = (half_t*)alloc(64 * 32 * 2);
    half_t* fq16    = (half_t*)alloc(64 * 32 * 2);
    half_t* fkvk16  = (half_t*)alloc(64 * 64 * 2);
    half_t* fkvv16  = (half_t*)alloc(64 * 64 * 2);
    half_t* head16  = (half_t*)alloc(16 * 128 * 2);
    float*  maxx    = (float*) alloc(BATCH * 4);
    float*  maxl    = (float*) alloc(BATCH * 4);
    float*  sumx    = (float*) alloc(BATCH * 4);
    float*  suml    = (float*) alloc(BATCH * 4);
    float*  scal    = (float*) alloc(16 * 4);

    auto gemm = [&](const half_t* A, const half_t* W16, const float* bias, const float* res,
                    float* C, int M, int Nreal, int Kpad, int lda, int ldc, int act,
                    const float* g, const float* bt) {
        int Npad = ((Nreal + 15) / 16) * 16;
        dim3 grid(M / 16, (Nreal + 63) / 64);
        gemm_wmma<<<grid, 32, 0, stream>>>(A, W16, bias, res, C, M, Nreal, Npad, Kpad, lda, ldc, act, g, bt);
    };
    auto padw = [&](const float* src, half_t* dst, int R, int C, int Rout, int Cout) {
        int tot = Rout * Cout;
        pad2d_f16<<<(tot + 255) / 256, 256, 0, stream>>>(src, dst, R, C, Rout, Cout);
    };

    init_scal<<<1, 32, 0, stream>>>(scal);

    // ----- stem: spectral conv -> h2 (f16), conv GEMM (K=1120), lidar convs -----
    {
        int tot = MSP * C2;
        conv3d_h2<<<(tot + 255) / 256, 256, 0, stream>>>(x, w3d, b3d, bn3d, h2f16, tot);
    }
    padw(convx_w, convxw16, 64, C2, 64, C2);
    gemm(h2f16, convxw16, convx_b, nullptr, xfeat, MSP, 64, C2, C2, 64, 1, bnx, bnx + 64);
    {
        int tot = MSP * 32;
        lidar_pack<<<(tot + 255) / 256, 256, 0, stream>>>(lidar, lidA16, tot);
    }
    padw(lidw1, lidw1_16, 32, 4, 32, 32);
    gemm(lidA16, lidw1_16, lidb1, nullptr, lf1, MSP, 32, 32, 32, 32, 1, bnl1, bnl1 + 32);
    pad2d_f16<<<(MSP * 32 + 255) / 256, 256, 0, stream>>>(lf1, lf1_16, MSP, 32, MSP, 32);
    padw(lidw2, lidw2_16, 64, 32, 64, 32);
    gemm(lf1_16, lidw2_16, lidb2, nullptr, lfeat, MSP, 64, 32, 32, 64, 1, bnl2, bnl2 + 64);
    build_tokens<<<(TOKENS * 64 + 255) / 256, 256, 0, stream>>>(xfeat, cls_tok, pos1, xtok, TOKENS * 64);
    build_tokens<<<(TOKENS * 64 + 255) / 256, 256, 0, stream>>>(lfeat, cls_tok, pos2, ltok, TOKENS * 64);

    // ----- encoder depth loop -----
    for (int dep = 0; dep < 2; ++dep) {
        for (int s = 0; s < 2; ++s) {
            float* tok = s ? ltok : xtok;
            float* qkv = s ? qkvl : qkvx;
            float* dts = s ? dots_l : dots_x;
            const float* lg = ln_att + ((dep * 2 + s) * 2 + 0) * 64;
            const float* lb = ln_att + ((dep * 2 + s) * 2 + 1) * 64;
            layernorm_f16<<<TOKENS, 64, 0, stream>>>(tok, lg, lb, ln16);
            padw(qkv_w + (size_t)(dep * 2 + s) * 192 * 64, qkvw16, 192, 64, 192, 64);
            gemm(ln16, qkvw16, qkv_b + (dep * 2 + s) * 192, nullptr, qkv, TOKENS, 192, 64, 64, 192, 0, nullptr, nullptr);
            attn_scores<<<dim3(8, 8, BATCH * HC), 32, 0, stream>>>(qkv, dts);
        }
        {
            int tot = BATCH * PERB;
            sup_mix<<<(tot + 255) / 256, 256, 0, stream>>>(dots_x, dots_l,
                aw1 + dep * 128, ab1 + dep * 64, abn + dep * 128, abn + dep * 128 + 64,
                aw2 + dep * 64, ab2 + dep * 8, tot);
        }
        kl_max<<<BATCH, 256, 0, stream>>>(dots_x, dots_l, maxx, maxl);
        kl_sumexp<<<BATCH, 256, 0, stream>>>(dots_x, dots_l, maxx, maxl, sumx, suml);
        kl_accum<<<BATCH, 256, 0, stream>>>(dots_x, dots_l, maxx, maxl, sumx, suml, scal);
        for (int s = 0; s < 2; ++s) {
            float* tok = s ? ltok : xtok;
            float* qkv = s ? qkvl : qkvx;
            float* dts = s ? dots_l : dots_x;
            softmax_row<<<BATCH * HC * NTOK, 32, 0, stream>>>(dts, P16);
            attn_pv<<<dim3(8, BATCH * HC), 32, 0, stream>>>(P16, qkv, ctx);
            pad2d_f16<<<(TOKENS * 64 + 255) / 256, 256, 0, stream>>>(ctx, ctx16, TOKENS, 64, TOKENS, 64);
            padw(enc_mw + (size_t)(dep * 2 + s) * 64 * 64, encw16, 64, 64, 64, 64);
            gemm(ctx16, encw16, enc_mb + (dep * 2 + s) * 64, tok, tok, TOKENS, 64, 64, 64, 64, 0, nullptr, nullptr);
        }
        for (int s = 0; s < 2; ++s) {
            float* tok = s ? ltok : xtok;
            const float* lg = ln_ff + ((dep * 2 + s) * 2 + 0) * 64;
            const float* lb = ln_ff + ((dep * 2 + s) * 2 + 1) * 64;
            layernorm_f16<<<TOKENS, 64, 0, stream>>>(tok, lg, lb, ln16);
            padw(ff_w1 + (size_t)(dep * 2 + s) * 8 * 64, ffw1_16, 8, 64, 16, 64);
            gemm(ln16, ffw1_16, ff_b1 + (dep * 2 + s) * 8, nullptr, hidden, TOKENS, 8, 64, 64, 8, 2, nullptr, nullptr);
            pad2d_f16<<<(TOKENS * 32 + 255) / 256, 256, 0, stream>>>(hidden, hid16, TOKENS, 8, TOKENS, 32);
            padw(ff_w2 + (size_t)(dep * 2 + s) * 64 * 8, ffw2_16, 64, 8, 64, 32);
            gemm(hid16, ffw2_16, ff_b2 + (dep * 2 + s) * 64, tok, tok, TOKENS, 64, 32, 32, 64, 0, nullptr, nullptr);
        }
    }

    // ----- prototype scoring, masking, fusion, head -----
    proto_scores<<<BATCH, 32, 0, stream>>>(xtok, ltok, x_proto, l_proto, score1, score2, s1p, s2p);
    pro_kernel<<<1, 128, 0, stream>>>(score1, score2, label, scal);
    mask_mul<<<(TOKENS * 64 + 255) / 256, 256, 0, stream>>>(xtok, rand1, scal, 0, xm, xm16, TOKENS * 64);
    mask_mul<<<(TOKENS * 64 + 255) / 256, 256, 0, stream>>>(ltok, rand2, scal, 1, lm, lm16, TOKENS * 64);

    for (int st = 0; st < 2; ++st) {
        half_t* sc16  = st ? s1p : s2p;            // stream1 uses score2, stream2 uses score1
        half_t* feat16= st ? lm16 : xm16;
        float*  qb    = st ? q2 : q1;
        float*  fb    = st ? f2 : f1;
        padw(fq_w + (size_t)st * 64 * NCLS, fq16, 64, NCLS, 64, 32);
        gemm(sc16, fq16, fq_b + st * 64, nullptr, qb, BATCH, 64, 32, 32, 64, 0, nullptr, nullptr);
        padw(fkv_w + (size_t)(st * 2 + 0) * 64 * 64, fkvk16, 64, 64, 64, 64);
        gemm(feat16, fkvk16, fkv_b + (st * 2 + 0) * 64, nullptr, kbuf, TOKENS, 64, 64, 64, 64, 0, nullptr, nullptr);
        padw(fkv_w + (size_t)(st * 2 + 1) * 64 * 64, fkvv16, 64, 64, 64, 64);
        gemm(feat16, fkvv16, fkv_b + (st * 2 + 1) * 64, nullptr, vbuf, TOKENS, 64, 64, 64, 64, 0, nullptr, nullptr);
        fuse_attn<<<BATCH * HC, 128, 0, stream>>>(qb, kbuf, vbuf, fb);
    }
    build_fused<<<(BATCH * 128 + 255) / 256, 256, 0, stream>>>(f1, f2, xm, lm, fused, fused16);
    padw(head_w, head16, NCLS, 128, 16, 128);
    gemm(fused16, head16, head_b, nullptr, (float*)d_out, BATCH, NCLS, 128, 128, NCLS, 0, nullptr, nullptr);
    copy_out<<<(16385 + 255) / 256, 256, 0, stream>>>(scal, xtok, ltok, (float*)d_out);
}
